// SemanticPropagationUpsample_90993177133309
// MI455X (gfx1250) — matrix operand order repaired and checked by reading, not compiled, not run on hardware
//
#include <hip/hip_runtime.h>
#include <math.h>

// ---- problem constants (match reference setup_inputs) ----
#define BB    8
#define CC    128
#define HH    64
#define WWI   64
#define NN    16384          // Hp*Wp = 1<<14
#define BN    131072         // BB*NN
#define CHUNK 128
#define NCH   128            // NN / CHUNK
#define MTILES (BN / 16)     // 8192

typedef __attribute__((ext_vector_type(16))) _Float16 v16h;
typedef __attribute__((ext_vector_type(8)))  _Float16 v8h;
typedef __attribute__((ext_vector_type(8)))  float    v8f;

// ----------------- WMMA fragment loaders (CDNA5 wave32 layouts) -----------------
// A-matrix 16x32 f16: lane&15 = row M; lane>=16 selects K halves {8..15,24..31}.
__device__ __forceinline__ v16h load_a_frag_g(const _Float16* xn, int rowGlob, int kb, int lane) {
    int khalf = lane >> 4;
    const _Float16* p0 = xn + (size_t)rowGlob * CC + kb * 32 + khalf * 8;
    v8h lo = *(const v8h*)(p0);
    v8h hi = *(const v8h*)(p0 + 16);
    v16h a;
#pragma unroll
    for (int i = 0; i < 8; ++i) { a[i] = lo[i]; a[i + 8] = hi[i]; }
    return a;
}

// B-matrix 32x16 f16: lane&15 = col N; lane>=16 selects K halves {8..15,24..31}. W stored (o,c) f16.
__device__ __forceinline__ v16h load_b_frag(const _Float16* W, int ocol, int kb, int lane) {
    int kbase = (lane >> 4) * 8;
    const _Float16* p = W + (size_t)ocol * CC + kb * 32 + kbase;
    v8h lo = *(const v8h*)(p);
    v8h hi = *(const v8h*)(p + 16);
    v16h b;
#pragma unroll
    for (int i = 0; i < 8; ++i) { b[i] = lo[i]; b[i + 8] = hi[i]; }
    return b;
}

#define WMMA_F16(a, b, c) \
    __builtin_amdgcn_wmma_f32_16x16x32_f16(false, (a), false, (b), (short)0, (c), false, false)

// ----------------- prep kernels -----------------
__global__ void spu_conv_weights(const float* Wk, const float* Wv, const float* Wr,
                                 const float* Wo, const float* gw, const float* bw,
                                 _Float16* wh) {
    int tid = blockIdx.x * blockDim.x + threadIdx.x;
    if (tid >= 12 * CC * CC) return;
    int mat = tid >> 14;
    int e   = tid & 16383;
    int blk = mat / 6, which = mat % 6;
    const float* src;
    switch (which) {
        case 0: src = Wk; break;
        case 1: src = Wv; break;
        case 2: src = Wr; break;
        case 3: src = Wo; break;
        case 4: src = gw; break;
        default: src = bw; break;
    }
    wh[tid] = (_Float16)src[blk * CC * CC + e];
}

__global__ void spu_film(const float* gs, const float* cgw, const float* cgb,
                         const float* cbw, const float* cbb,
                         float* gamma, float* beta) {
    int tid = blockIdx.x * blockDim.x + threadIdx.x;
    if (tid >= BB * CC) return;
    int b = tid >> 7, o = tid & 127;
    float sg = 0.f, sb = 0.f;
#pragma unroll 4
    for (int c = 0; c < CC; ++c) {
        float xv = gs[b * CC + c];
        sg += xv * cgw[o * CC + c];
        sb += xv * cbw[o * CC + c];
    }
    gamma[tid] = sg + cgb[o];
    beta[tid]  = sb + cbb[o];
}

// Bilinear 2x upsample (half-pixel centers, edge clamp) -> d_out (B,C,Hp,Wp)
__global__ void spu_upsample(const float* __restrict__ x, float* __restrict__ out) {
    int tid = blockIdx.x * blockDim.x + threadIdx.x;
    if (tid >= BB * CC * NN) return;
    int n  = tid & (NN - 1);
    int bc = tid >> 14;
    int h = n >> 7, w = n & 127;
    int jy = h >> 1, jx = w >> 1;
    int y0, y1, x0, x1; float wy0, wx0;
    if (h & 1) { y0 = jy; y1 = (jy + 1 < HH) ? jy + 1 : HH - 1; wy0 = 0.75f; }
    else       { y0 = (jy - 1 >= 0) ? jy - 1 : 0; y1 = jy; wy0 = 0.25f; }
    if (w & 1) { x0 = jx; x1 = (jx + 1 < WWI) ? jx + 1 : WWI - 1; wx0 = 0.75f; }
    else       { x0 = (jx - 1 >= 0) ? jx - 1 : 0; x1 = jx; wx0 = 0.25f; }
    const float* xp = x + (size_t)bc * (HH * WWI);
    float r0 = wx0 * xp[y0 * WWI + x0] + (1.f - wx0) * xp[y0 * WWI + x1];
    float r1 = wx0 * xp[y1 * WWI + x0] + (1.f - wx0) * xp[y1 * WWI + x1];
    out[tid] = wy0 * r0 + (1.f - wy0) * r1;
}

// ----------------- fused LayerNorm(+FiLM) + k/v/r GEMM -----------------
// One block per 16-token tile. Async-DMA the f32 tile from (B,C,N) `src` into
// LDS, layernorm it, build the f16 A-tile in LDS, then 8 waves x 3 WMMA GEMMs.
__global__ void spu_ln_kvr(const float* __restrict__ src,
                           const float* __restrict__ gamma, const float* __restrict__ beta,
                           const float* __restrict__ lng, const float* __restrict__ lnb,
                           const _Float16* __restrict__ wh,
                           _Float16* __restrict__ kh, _Float16* __restrict__ vh,
                           _Float16* __restrict__ rh, int use_film) {
    __shared__ float    raw[CC * 16];   // [c][tok]  8 KB (matches global contiguity)
    __shared__ _Float16 xnl[16 * CC];   // [tok][c]  4 KB (A-tile layout)
    int tid  = threadIdx.x;
    int wave = tid >> 5, lane = tid & 31;
    int mtile = blockIdx.x;
    int tok0 = mtile * 16;
    int b = tok0 >> 14, n0 = tok0 & (NN - 1);

    // ---- async stage: each thread DMAs 8 consecutive-n floats of one channel ----
    {
        int c = tid >> 1, half = tid & 1;
        const float* g0 = src + ((size_t)(b * CC + c) << 14) + n0 + half * 8;
        unsigned l0 = (unsigned)(uintptr_t)&raw[c * 16 + half * 8];
        asm volatile(
            "global_load_async_to_lds_b128 %0, %1, off\n\t"
            "global_load_async_to_lds_b128 %0, %1, off offset:16"
            :: "v"(l0), "v"(g0) : "memory");
        asm volatile("s_wait_asynccnt 0x0" ::: "memory");
    }
    __syncthreads();

    // ---- LayerNorm (+FiLM): half-wave per token, 8 channels per lane ----
    {
        int halfw = lane >> 4, li = lane & 15;
        int t = wave * 2 + halfw;           // token within tile
        float vals[8];
        float s1 = 0.f, s2 = 0.f;
#pragma unroll
        for (int j = 0; j < 8; ++j) {
            int c = li + j * 16;
            float v = raw[c * 16 + t];
            if (use_film) v = v * (1.f + gamma[b * CC + c]) + beta[b * CC + c];
            vals[j] = v; s1 += v; s2 += v * v;
        }
#pragma unroll
        for (int off = 8; off >= 1; off >>= 1) {
            s1 += __shfl_xor(s1, off, 16);
            s2 += __shfl_xor(s2, off, 16);
        }
        float mu  = s1 * (1.f / CC);
        float var = s2 * (1.f / CC) - mu * mu;
        float rs  = rsqrtf(var + 1e-5f);
#pragma unroll
        for (int j = 0; j < 8; ++j) {
            int c = li + j * 16;
            xnl[t * CC + c] = (_Float16)((vals[j] - mu) * rs * lng[c] + lnb[c]);
        }
    }
    __syncthreads();

    // ---- WMMA: wave = ntile; A fragments from LDS, B (weights) from global ----
    int ntile = wave;
    int rowl = lane & 15, khalf = lane >> 4;
    v16h a[4];
#pragma unroll
    for (int kb = 0; kb < 4; ++kb) {
        const _Float16* p0 = &xnl[rowl * CC + kb * 32 + khalf * 8];
        v8h lo = *(const v8h*)(p0);
        v8h hi = *(const v8h*)(p0 + 16);
        v16h af;
#pragma unroll
        for (int i = 0; i < 8; ++i) { af[i] = lo[i]; af[i + 8] = hi[i]; }
        a[kb] = af;
    }
    int ocol  = ntile * 16 + rowl;
    int mbase = khalf * 8;
#pragma unroll
    for (int j = 0; j < 3; ++j) {
        const _Float16* W = wh + (size_t)j * CC * CC;
        v8f acc = {};
#pragma unroll
        for (int kb = 0; kb < 4; ++kb) {
            v16h bf = load_b_frag(W, ocol, kb, lane);
            acc = WMMA_F16(a[kb], bf, acc);
        }
        _Float16* dst = (j == 0) ? kh : ((j == 1) ? vh : rh);
#pragma unroll
        for (int i = 0; i < 8; ++i) {
            float vv = acc[i];
            if (j == 2) vv = 1.f / (1.f + __expf(-vv));   // sigmoid(r)
            dst[(size_t)(tok0 + mbase + i) * CC + ocol] = (_Float16)vv;
        }
    }
}

// ----------------- fused t = y@Wo^T -> LDS -> {g,bt} GEMMs -> gate update -----------------
__global__ void spu_wo_gate(const _Float16* __restrict__ yh, const _Float16* __restrict__ Wo,
                            const _Float16* __restrict__ Wg, const _Float16* __restrict__ Wb,
                            const float* __restrict__ gpb, const float* __restrict__ bpb,
                            float* __restrict__ out) {
    __shared__ _Float16 tl[16 * CC];    // full 16x128 t-tile, 4 KB
    int wave = threadIdx.x >> 5, lane = threadIdx.x & 31;
    int mtile = blockIdx.x, ntile = wave;
    int row0 = mtile * 16;
    int rowl = lane & 15, khalf = lane >> 4;
    int row  = row0 + rowl;
    int ocol  = ntile * 16 + rowl;
    int mbase = khalf * 8;

    // t = y @ Wo^T (this wave's 16-column slab); 8 waves cover all 128 columns
    v8f acc = {};
#pragma unroll
    for (int kb = 0; kb < 4; ++kb) {
        v16h af = load_a_frag_g(yh, row, kb, lane);
        v16h bf = load_b_frag(Wo, ocol, kb, lane);
        acc = WMMA_F16(af, bf, acc);
    }
#pragma unroll
    for (int i = 0; i < 8; ++i)
        tl[(mbase + i) * CC + ocol] = (_Float16)acc[i];
    __syncthreads();

    // A2 fragments from the LDS t-tile
    v16h a[4];
#pragma unroll
    for (int kb = 0; kb < 4; ++kb) {
        const _Float16* p0 = &tl[rowl * CC + kb * 32 + khalf * 8];
        v8h lo = *(const v8h*)(p0);
        v8h hi = *(const v8h*)(p0 + 16);
        v16h af;
#pragma unroll
        for (int i = 0; i < 8; ++i) { af[i] = lo[i]; af[i + 8] = hi[i]; }
        a[kb] = af;
    }
    v8f accG = {}, accB = {};
#pragma unroll
    for (int kb = 0; kb < 4; ++kb) {
        v16h bg = load_b_frag(Wg, ocol, kb, lane);
        accG = WMMA_F16(a[kb], bg, accG);
    }
#pragma unroll
    for (int kb = 0; kb < 4; ++kb) {
        v16h bb = load_b_frag(Wb, ocol, kb, lane);
        accB = WMMA_F16(a[kb], bb, accB);
    }
    float bg = gpb[ocol], bbias = bpb[ocol];
#pragma unroll
    for (int i = 0; i < 8; ++i) {
        int tok = row0 + mbase + i;
        int b = tok >> 14, n = tok & (NN - 1);
        size_t oidx = ((size_t)(b * CC + ocol) << 14) + n;
        float ov = out[oidx];
        float g  = tanhf(accG[i] + bg);
        out[oidx] = ov * (1.f + g) + (accB[i] + bbias);
    }
}

// ----------------- chunked linear-recurrence scan -----------------
__global__ void spu_scan1(const _Float16* __restrict__ kh, const float* __restrict__ td,
                          float* __restrict__ carry) {
    int tid = blockIdx.x * blockDim.x + threadIdx.x;   // B*NCH*C = 131072
    if (tid >= BB * NCH * CC) return;
    int c = tid & 127;
    int bj = tid >> 7;
    int chunk = bj & (NCH - 1);
    int b = bj >> 7;
    float dec = __expf(-fmaxf(td[c], 0.f));
    const _Float16* kp = kh + ((size_t)(b * NN + chunk * CHUNK)) * CC + c;
    float s = 0.f;
#pragma unroll 4
    for (int i = 0; i < CHUNK; ++i) s = s * dec + (float)kp[(size_t)i * CC];
    carry[tid] = s;
}

__global__ void spu_scan2(const float* __restrict__ carry, const float* __restrict__ td,
                          float* __restrict__ init) {
    int tid = blockIdx.x * blockDim.x + threadIdx.x;   // B*C = 1024
    if (tid >= BB * CC) return;
    int c = tid & 127, b = tid >> 7;
    float dec = __expf(-fmaxf(td[c], 0.f));
    float dl = dec;
#pragma unroll
    for (int q = 0; q < 7; ++q) dl *= dl;              // dec^128
    float s = 0.f;
    for (int j = 0; j < NCH; ++j) {
        int idx = ((b * NCH + j) << 7) + c;
        init[idx] = s;
        s = s * dl + carry[idx];
    }
}

__global__ void spu_scan3(const _Float16* __restrict__ kh, const _Float16* __restrict__ vh,
                          const _Float16* __restrict__ rh, const float* __restrict__ init,
                          const float* __restrict__ td, _Float16* __restrict__ yh) {
    int tid = blockIdx.x * blockDim.x + threadIdx.x;
    if (tid >= BB * NCH * CC) return;
    int c = tid & 127;
    int bj = tid >> 7;
    int chunk = bj & (NCH - 1);
    int b = bj >> 7;
    float dec = __expf(-fmaxf(td[c], 0.f));
    size_t off = ((size_t)(b * NN + chunk * CHUNK)) * CC + c;
    float s = init[tid];
#pragma unroll 4
    for (int i = 0; i < CHUNK; ++i) {
        size_t o = off + (size_t)i * CC;
        s = s * dec + (float)kh[o];
        float y = (float)rh[o] * (s + (float)vh[o]);
        yh[o] = (_Float16)y;
    }
}

// ----------------- host launch -----------------
extern "C" void kernel_launch(void* const* d_in, const int* in_sizes, int n_in,
                              void* d_out, int out_size, void* d_ws, size_t ws_size,
                              hipStream_t stream) {
    (void)in_sizes; (void)n_in; (void)out_size; (void)ws_size;
    const float* x    = (const float*)d_in[0];
    const float* gs   = (const float*)d_in[1];
    const float* cgw  = (const float*)d_in[2];
    const float* cgb  = (const float*)d_in[3];
    const float* cbw  = (const float*)d_in[4];
    const float* cbb  = (const float*)d_in[5];
    const float* td   = (const float*)d_in[6];
    const float* Wk   = (const float*)d_in[7];
    const float* Wv   = (const float*)d_in[8];
    const float* Wr   = (const float*)d_in[9];
    const float* Wo   = (const float*)d_in[10];
    const float* lng  = (const float*)d_in[11];
    const float* lnb  = (const float*)d_in[12];
    const float* gpw  = (const float*)d_in[13];
    const float* gpb  = (const float*)d_in[14];
    const float* bpw  = (const float*)d_in[15];
    const float* bpb  = (const float*)d_in[16];
    float* out = (float*)d_out;

    char* ws = (char*)d_ws;
    const size_t T16 = (size_t)BN * CC * sizeof(_Float16);   // 33,554,432 B
    _Float16* wh    = (_Float16*)(ws + 0);                   // 393,216 B
    float*    gamma = (float*)(ws + 393216);
    float*    beta  = (float*)(ws + 397312);
    float*    carry = (float*)(ws + 401408);                 // 512 KB
    float*    init  = (float*)(ws + 925696);                 // 512 KB
    _Float16* kh    = (_Float16*)(ws + 2097152);
    _Float16* vh    = (_Float16*)(ws + 2097152 + 1 * T16);
    _Float16* rh    = (_Float16*)(ws + 2097152 + 2 * T16);
    _Float16* yh    = (_Float16*)(ws + 2097152 + 3 * T16);   // ends ~136 MB

    spu_conv_weights<<<(12 * CC * CC + 255) / 256, 256, 0, stream>>>(Wk, Wv, Wr, Wo, gpw, bpw, wh);
    spu_film<<<(BB * CC + 255) / 256, 256, 0, stream>>>(gs, cgw, cgb, cbw, cbb, gamma, beta);
    spu_upsample<<<(BB * CC * NN + 255) / 256, 256, 0, stream>>>(x, out);

    for (int blk = 0; blk < 2; ++blk) {
        const _Float16* whB  = wh + (size_t)blk * 6 * CC * CC;  // k,v,r consecutive
        const _Float16* Wo_h = whB + 3 * CC * CC;
        const _Float16* Wg_h = whB + 4 * CC * CC;
        const _Float16* Wb_h = whB + 5 * CC * CC;
        const float* tdB  = td  + blk * CC;
        const float* lngB = lng + blk * CC;
        const float* lnbB = lnb + blk * CC;
        const float* gpbB = gpb + blk * CC;
        const float* bpbB = bpb + blk * CC;

        // fused LN(+FiLM) + k/v/sigmoid(r) GEMMs
        spu_ln_kvr<<<MTILES, 256, 0, stream>>>(out, gamma, beta, lngB, lnbB, whB,
                                               kh, vh, rh, blk == 0 ? 1 : 0);
        // chunked scan -> y
        spu_scan1<<<(BB * NCH * CC + 255) / 256, 256, 0, stream>>>(kh, tdB, carry);
        spu_scan2<<<(BB * CC + 255) / 256, 256, 0, stream>>>(carry, tdB, init);
        spu_scan3<<<(BB * NCH * CC + 255) / 256, 256, 0, stream>>>(kh, vh, rh, init, tdB, yh);
        // fused Wo GEMM -> LDS t-tile -> gate GEMMs -> out update
        spu_wo_gate<<<MTILES, 256, 0, stream>>>(yh, Wo_h, Wg_h, Wb_h, gpbB, bpbB, out);
    }
}
